// RGB_SFMCNN_32126355374717
// MI455X (gfx1250) — matrix operand order repaired and checked by reading, not compiled, not run on hardware
//
#include <hip/hip_runtime.h>
#include <math.h>

typedef __attribute__((ext_vector_type(16))) __bf16 v16bf;
typedef __attribute__((ext_vector_type(8)))  __bf16 v8bf;
typedef __attribute__((ext_vector_type(8)))  float  v8f;

#define N_      32
#define C_      150
#define H_      56
#define W_      56
#define HW_     (H_*W_)          // 3136
#define O_      225
#define OPAD    240              // 15 tiles of 16
#define D_      1350
#define DPAD    1376             // 43 * 32, zero padded
#define NCHUNK  43
#define SPLITD  675              // SPLIT(75) * 9
#define NROWS   (N_*HW_)         // 100352 = 784 * 128
#define MBLK    128
#define NT      15
#define ASTRIDE 40               // bf16 elements per A row (80B, conflict-free, 16B aligned)
#define BSTRIDE 40

#define CAT16(a,b) __builtin_shufflevector((a),(b),0,1,2,3,4,5,6,7,8,9,10,11,12,13,14,15)

// ---------------- prep: weight * wfull, split into bf16 hi/lo, (OPAD x DPAD) ----------------
__global__ __launch_bounds__(256) void sfm_prep_w(const float* __restrict__ w,
                                                  __bf16* __restrict__ wph,
                                                  __bf16* __restrict__ wpl) {
    int idx = blockIdx.x * 256 + threadIdx.x;        // over OPAD*DPAD = 330240
    if (idx >= OPAD * DPAD) return;
    int o = idx / DPAD;
    int d = idx - o * DPAD;
    float v = 0.0f;
    if (o < O_ && d < D_)
        v = w[o * D_ + d] * (d < SPLITD ? 0.25f : 0.75f);
    __bf16 hi = (__bf16)v;
    __bf16 lo = (__bf16)(v - (float)hi);
    wph[idx] = hi;
    wpl[idx] = lo;
}

// ---------------- prep: p2[o] = sum_d wfull_d * p_od^2 (padded to OPAD with zeros) ----------
__global__ __launch_bounds__(256) void sfm_prep_p2(const float* __restrict__ w,
                                                   float* __restrict__ p2) {
    int o = threadIdx.x;
    if (o >= OPAD) return;
    float s = 0.0f;
    if (o < O_) {
        const float* row = w + o * D_;
        for (int d = 0; d < D_; ++d) {
            float v = row[d];
            s += (d < SPLITD ? 0.25f : 0.75f) * v * v;
        }
    }
    p2[o] = s;
}

// ---------------- prep: y[n,h,w] = sum_c w_c * x^2 (reads x exactly once) -------------------
__global__ __launch_bounds__(256) void sfm_prep_y(const float* __restrict__ x,
                                                  float* __restrict__ y) {
    int idx = blockIdx.x * 256 + threadIdx.x;        // < NROWS
    if (idx >= NROWS) return;
    int n   = idx / HW_;
    int rem = idx - n * HW_;
    const float* p = x + n * (C_ * HW_) + rem;
    float s = 0.0f;
    for (int c = 0; c < C_; ++c) {
        float v = p[c * HW_];
        s += (c < 75 ? 0.25f : 0.75f) * v * v;
    }
    y[idx] = s;
}

// ---------------- prep: x2[row] = 3x3 zero-padded box sum of y ------------------------------
__global__ __launch_bounds__(256) void sfm_prep_x2(const float* __restrict__ y,
                                                   float* __restrict__ x2) {
    int idx = blockIdx.x * 256 + threadIdx.x;
    if (idx >= NROWS) return;
    int n   = idx / HW_;
    int rem = idx - n * HW_;
    int h   = rem / W_;
    int w   = rem - h * W_;
    const float* yn = y + n * HW_;
    float s = 0.0f;
    #pragma unroll
    for (int dh = -1; dh <= 1; ++dh) {
        #pragma unroll
        for (int dw = -1; dw <= 1; ++dw) {
            int hh = h + dh, ww = w + dw;
            if ((unsigned)hh < (unsigned)H_ && (unsigned)ww < (unsigned)W_)
                s += yn[hh * W_ + ww];
        }
    }
    x2[idx] = s;
}

// ---------------- main: 128 rows x 240 protos per block, bf16x3 split WMMA GEMM,
// ---------------- software-pipelined: chunk ch+1 global loads overlap chunk ch WMMAs -------
__global__ __launch_bounds__(256) void sfm_wmma_main(
    const float*  __restrict__ x,
    const __bf16* __restrict__ wph,
    const __bf16* __restrict__ wpl,
    const float*  __restrict__ p2,
    const float*  __restrict__ x2,
    float*        __restrict__ out)
{
    __shared__ __align__(16) char smem[(MBLK*ASTRIDE*2 + OPAD*BSTRIDE*2) * 2]; // 58,880 B
    __bf16* Ah = (__bf16*)smem;
    __bf16* Al = Ah + MBLK * ASTRIDE;
    __bf16* Bh = Al + MBLK * ASTRIDE;
    __bf16* Bl = Bh + OPAD * BSTRIDE;
    float*  Cs = (float*)smem;                       // reused after the K loop

    const int tid   = threadIdx.x;
    const int wave  = tid >> 5;
    const int lane  = tid & 31;
    const int khalf = lane >> 4;                     // 0 / 1 half-wave
    const int l15   = lane & 15;

    // ---- per-thread im2col coordinates for A staging (m fixed per thread) ----
    const int ms   = tid & 127;                      // staged row within block
    const int kk0  = (tid >> 7) * 16;                // staged k sub-range (0 or 16)
    const int rowS = blockIdx.x * MBLK + ms;
    const int nS   = rowS / HW_;
    const int remS = rowS - nS * HW_;
    const int hS   = remS / W_;
    const int wS   = remS - hS * W_;
    const float* xn = x + nS * (C_ * HW_);

    // ---- register staging buffers (double-buffer vs LDS) ----
    float av[16];
    v8bf  bvh[4], bvl[4];

    auto loadA = [&](int cc) {                       // im2col gather for chunk cc -> regs
        #pragma unroll
        for (int j = 0; j < 16; ++j) {
            int kk = kk0 + j;
            int d  = cc * 32 + kk;
            float v = 0.0f;
            if (d < D_) {
                int c  = d / 9;
                int r9 = d - c * 9;
                int kh = r9 / 3;
                int kw = r9 - kh * 3;
                int hh = hS + kh - 1;
                int ww = wS + kw - 1;
                if ((unsigned)hh < (unsigned)H_ && (unsigned)ww < (unsigned)W_)
                    v = xn[c * HW_ + hh * W_ + ww];
            }
            av[j] = v;
        }
    };
    auto loadB = [&](int cc) {                       // pre-split weights for chunk cc -> regs
        #pragma unroll
        for (int j = 0; j < 4; ++j) {
            int e = tid + 256 * j;                   // OPAD*4 = 960 vec8 moves
            if (e < OPAD * 4) {
                int o   = e >> 2;
                int seg = e & 3;
                int g   = o * DPAD + cc * 32 + seg * 8;
                bvh[j] = *(const v8bf*)(wph + g);
                bvl[j] = *(const v8bf*)(wpl + g);
                if (cc + 1 < NCHUNK) {               // prefetch one more chunk ahead
                    __builtin_prefetch(wph + g + 32, 0, 1);
                    __builtin_prefetch(wpl + g + 32, 0, 1);
                }
            }
        }
    };
    auto storeAB = [&]() {                           // regs -> LDS (bf16 split for A)
        #pragma unroll
        for (int j = 0; j < 16; ++j) {
            int kk = kk0 + j;
            float v = av[j];
            __bf16 hi = (__bf16)v;
            __bf16 lo = (__bf16)(v - (float)hi);
            Ah[ms * ASTRIDE + kk] = hi;
            Al[ms * ASTRIDE + kk] = lo;
        }
        #pragma unroll
        for (int j = 0; j < 4; ++j) {
            int e = tid + 256 * j;
            if (e < OPAD * 4) {
                int o   = e >> 2;
                int seg = e & 3;
                int s   = o * BSTRIDE + seg * 8;
                *(v8bf*)(Bh + s) = bvh[j];
                *(v8bf*)(Bl + s) = bvl[j];
            }
        }
    };

    v8f acc[NT] = {};

    loadA(0);
    loadB(0);

    for (int ch = 0; ch < NCHUNK; ++ch) {
        __syncthreads();                             // previous chunk's compute done
        storeAB();
        __syncthreads();                             // tiles visible to all waves

        if (ch + 1 < NCHUNK) {                       // issue next chunk's global loads now;
            loadA(ch + 1);                           // latency hidden behind 45 WMMAs below
            loadB(ch + 1);
        }

        // ---- A fragments (ISA 16-bit A 16x32 layout: regs0-3 K=khalf*8+0..7, regs4-7 +16) --
        int aoff = (wave * 16 + l15) * ASTRIDE + khalf * 8;
        v8bf a0h = *(const v8bf*)(Ah + aoff);
        v8bf a1h = *(const v8bf*)(Ah + aoff + 16);
        v8bf a0l = *(const v8bf*)(Al + aoff);
        v8bf a1l = *(const v8bf*)(Al + aoff + 16);
        v16bf ah = CAT16(a0h, a1h);
        v16bf al = CAT16(a0l, a1l);

        // ---- B fragments + 3-term split WMMA per N-tile ----
        #pragma unroll
        for (int t = 0; t < NT; ++t) {
            int boff = (t * 16 + l15) * BSTRIDE + khalf * 8;   // K=khalf*8+0..7, then +16
            v8bf b0h = *(const v8bf*)(Bh + boff);
            v8bf b1h = *(const v8bf*)(Bh + boff + 16);
            v8bf b0l = *(const v8bf*)(Bl + boff);
            v8bf b1l = *(const v8bf*)(Bl + boff + 16);
            v16bf bh = CAT16(b0h, b1h);
            v16bf bl = CAT16(b0l, b1l);
            acc[t] = __builtin_amdgcn_wmma_f32_16x16x32_bf16(
                false, ah, false, bh, (short)0, acc[t], false, false);
            acc[t] = __builtin_amdgcn_wmma_f32_16x16x32_bf16(
                false, ah, false, bl, (short)0, acc[t], false, false);
            acc[t] = __builtin_amdgcn_wmma_f32_16x16x32_bf16(
                false, al, false, bh, (short)0, acc[t], false, false);
        }
    }

    __syncthreads();   // before aliasing smem as the per-wave transpose scratch

    // ---- epilogue: per-wave LDS transpose, fuse sqrt(x2 + p2 - 2*xp), coalesced store ----
    float* cs = Cs + wave * (16 * 17);
    const int row2 = blockIdx.x * MBLK + wave * 16 + l15;
    const int n2   = row2 / HW_;
    const int rem2 = row2 - n2 * HW_;
    const float xv = x2[row2];
    float* outp = out + n2 * (O_ * HW_) + rem2;

    for (int t = 0; t < NT; ++t) {
        #pragma unroll
        for (int r = 0; r < 8; ++r)                  // C layout: reg r -> M = r + 8*khalf
            cs[(r + 8 * khalf) * 17 + l15] = acc[t][r];
        __syncthreads();
        #pragma unroll
        for (int jj = 0; jj < 8; ++jj) {
            int oo = khalf + 2 * jj;                 // transposed read: lane = row m
            int og = t * 16 + oo;
            float cv  = cs[l15 * 17 + oo];
            float d2v = xv + p2[og] - 2.0f * cv;
            float dist = sqrtf(fmaxf(d2v, 1e-12f));
            if (og < O_)
                outp[og * HW_] = dist;               // 16 consecutive w per half-wave
        }
        __syncthreads();
    }
}

// ---------------- host-side launch ----------------
extern "C" void kernel_launch(void* const* d_in, const int* in_sizes, int n_in,
                              void* d_out, int out_size, void* d_ws, size_t ws_size,
                              hipStream_t stream) {
    const float* x      = (const float*)d_in[0];   // (32,150,56,56)
    const float* weight = (const float*)d_in[1];   // (225,1350)
    float*       out    = (float*)d_out;           // (32,225,56,56)

    char* ws = (char*)d_ws;
    __bf16* wph = (__bf16*)(ws);                        // 240*1376*2 = 660480 B
    __bf16* wpl = (__bf16*)(ws + 660480);               // 660480 B
    float*  p2  = (float*) (ws + 1320960);              // 1024 B (240 floats)
    float*  y   = (float*) (ws + 1321984);              // 401408 B
    float*  x2  = (float*) (ws + 1723392);              // 401408 B  (total ~2.03 MB)

    sfm_prep_w <<<(OPAD * DPAD) / 256, 256, 0, stream>>>(weight, wph, wpl);
    sfm_prep_p2<<<1, 256, 0, stream>>>(weight, p2);
    sfm_prep_y <<<NROWS / 256, 256, 0, stream>>>(x, y);
    sfm_prep_x2<<<NROWS / 256, 256, 0, stream>>>(y, x2);
    sfm_wmma_main<<<NROWS / MBLK, 256, 0, stream>>>(x, wph, wpl, p2, x2, out);
}
